// TemporalLSTM_67697274519650
// MI455X (gfx1250) — matrix operand order repaired and checked by reading, not compiled, not run on hardware
//
#include <hip/hip_runtime.h>
#include <hip/hip_bf16.h>

typedef __attribute__((ext_vector_type(16))) _Float16 v16h;
typedef __attribute__((ext_vector_type(8)))  _Float16 v8h;
typedef __attribute__((ext_vector_type(8)))  float    v8f;

#define B_   4096
#define S_   96
#define I_   64
#define HD_  128
#define T_   24
#define N_   64
#define BT   16      // batches per workgroup (= WMMA M)
#define NG   512     // 4*HD
#define KX   192     // I + HD  (gates GEMM K)

// ---------------- fast math (gfx1250 hardware transcendentals) ----------------

__device__ __forceinline__ float fast_tanh(float x) {
#if __has_builtin(__builtin_amdgcn_tanhf)
  return __builtin_amdgcn_tanhf(x);       // v_tanh_f32
#else
  return tanhf(x);
#endif
}

__device__ __forceinline__ float sigm(float x) {
  // sigmoid(x) = 0.5 * (1 + tanh(x/2)) -> single v_tanh_f32 + fma
  return fmaf(fast_tanh(x * 0.5f), 0.5f, 0.5f);
}

// ---------------- WMMA helpers (gfx1250 wave32 layouts) ----------------

// A-matrix 16x32 f16 per-lane load from row-major memory.
// p must already point at (row, kb) where kb = (lane<16)?0:8.
// Per ISA: lanes 0-15 elems 0..7 = K0..7, elems 8..15 = K16..23;
//          lanes 16-31 elems 0..7 = K8..15, elems 8..15 = K24..31.
__device__ __forceinline__ v16h ld_a16(const _Float16* p) {
  v8h lo = *(const v8h*)(p);
  v8h hi = *(const v8h*)(p + 16);
  v16h r;
#pragma unroll
  for (int j = 0; j < 8; ++j) { r[j] = lo[j]; r[j + 8] = hi[j]; }
  return r;
}

// B-matrix: pre-packed so each lane's 16 halves are contiguous (32B).
__device__ __forceinline__ v16h ld_b16(const _Float16* p) {
  return *(const v16h*)p;
}

__device__ __forceinline__ v8f wmma16(v16h a, v16h b, v8f c) {
  return __builtin_amdgcn_wmma_f32_16x16x32_f16(
      /*neg_a=*/false, a, /*neg_b=*/false, b,
      /*c_mod=*/(short)0, c, /*reuse_a=*/false, /*reuse_b=*/false);
}

// ---------------- pre-pass kernels ----------------

__global__ void convert_h_kernel(const float* __restrict__ H,
                                 _Float16* __restrict__ Hh, int n) {
  for (int i = blockIdx.x * blockDim.x + threadIdx.x; i < n;
       i += gridDim.x * blockDim.x)
    Hh[i] = (_Float16)H[i];
}

// B-pack element mapping: dst element e -> (block, lane, j);
// within a 32x16 (KxN) tile: n_rel = lane&15, k_rel = (j&8)*2 + ((lane>>4)&1)*8 + (j&7).
__global__ void pack_wa_kernel(const float* __restrict__ Wa,
                               _Float16* __restrict__ WaP) {
  int e = blockIdx.x * blockDim.x + threadIdx.x;
  if (e >= T_ * 2 * 4 * 512) return;               // 24 t * 2 kc * 4 nt * 512
  int j = e & 15, lane = (e >> 4) & 31, blk = e >> 9;
  int n_rel = lane & 15;
  int k_rel = (j & 8) * 2 + ((lane >> 4) & 1) * 8 + (j & 7);
  int nt = blk & 3, kc = (blk >> 2) & 1, t = blk >> 3;
  int k = kc * 32 + k_rel, n = nt * 16 + n_rel;
  WaP[e] = (_Float16)Wa[k * (T_ * N_) + t * N_ + n];   // Wa is [I, T, N]
}

__global__ void pack_ua_kernel(const float* __restrict__ Ua,
                               _Float16* __restrict__ UaP) {
  int e = blockIdx.x * blockDim.x + threadIdx.x;
  if (e >= T_ * 4 * 4 * 512) return;               // 24 t * 4 kc * 4 nt * 512
  int j = e & 15, lane = (e >> 4) & 31, blk = e >> 9;
  int n_rel = lane & 15;
  int k_rel = (j & 8) * 2 + ((lane >> 4) & 1) * 8 + (j & 7);
  int nt = blk & 3, kc = (blk >> 2) & 3, t = blk >> 4;
  int k = kc * 32 + k_rel, n = nt * 16 + n_rel;
  UaP[e] = (_Float16)Ua[k * (T_ * N_) + t * N_ + n];   // Ua is [HD, T, N]
}

__global__ void pack_wu_kernel(const float* __restrict__ W,
                               const float* __restrict__ U,
                               _Float16* __restrict__ WUP) {
  int e = blockIdx.x * blockDim.x + threadIdx.x;
  if (e >= 6 * 32 * 512) return;                   // 6 kc * 32 nt * 512
  int j = e & 15, lane = (e >> 4) & 31, blk = e >> 9;
  int n_rel = lane & 15;
  int k_rel = (j & 8) * 2 + ((lane >> 4) & 1) * 8 + (j & 7);
  int nt = blk & 31, kc = blk >> 5;
  int k = kc * 32 + k_rel, n = nt * 16 + n_rel;
  float v = (k < I_) ? W[k * NG + n]               // W rows 0..63 (ctx part)
                     : U[(k - I_) * NG + n];       // U rows 0..127 (h part)
  WUP[e] = (_Float16)v;
}

// ---------------- main fused scan kernel ----------------

__global__ __launch_bounds__(512)
void temporal_lstm_kernel(const float* __restrict__ H,      // [B,S,I] f32
                          const float* __restrict__ y0,     // [B,1]
                          const float* __restrict__ h0,     // [B,HD]
                          const float* __restrict__ c0,     // [B,HD]
                          const float* __restrict__ ba,     // [T,N]
                          const float* __restrict__ Va,     // [N,T,1]
                          const float* __restrict__ W,      // [I+1,4HD] (row 64 = y coeff)
                          const float* __restrict__ bias,   // [4HD]
                          const float* __restrict__ fcw,    // [T,HD]
                          const float* __restrict__ fcb,    // [T]
                          const _Float16* __restrict__ Hh,  // [B,S,I] f16
                          const _Float16* __restrict__ WaP, // packed B tiles
                          const _Float16* __restrict__ UaP,
                          const _Float16* __restrict__ WUP,
                          float* __restrict__ y_out,        // [B,T]
                          float* __restrict__ hid_out)      // [B,T,HD]
{
  __shared__ float    c32[BT][HD_];     // master cell state (f32)
  __shared__ _Float16 c16[BT][HD_];     // f16 copy for WMMA A
  __shared__ _Float16 xh16[BT][KX];     // [ctx(0:64) | h(64:192)] f16 A operand
  __shared__ float    cu[BT][N_];       // c @ Ua_t
  __shared__ float    e_l[BT][S_];      // logits -> beta (in place)
  __shared__ float    gates[BT][NG];
  __shared__ float    ystate[BT];
  __shared__ float    yacc[BT];

  const int tid  = threadIdx.x;
  const int lane = tid & 31;
  const int w    = tid >> 5;            // wave id, 0..15
  const int b0   = blockIdx.x * BT;     // global batch base
  const float SCALE = 0.0883883476483f; // 1/sqrt(128)

  // loop-invariant per-thread bases (hoisted to limit spills in the t-loop)
  const _Float16* HhWg   = Hh + (size_t)b0 * S_ * I_ + (size_t)(lane & 15) * I_;
  const float*    HWg    = H + (size_t)b0 * S_ * I_;
  float*          hidWg  = hid_out + (size_t)b0 * T_ * HD_;
  const float*    WyRow  = W + (size_t)I_ * NG;     // row 64 of W

  // ---- init state ----
  for (int idx = tid; idx < BT * HD_; idx += 512) {
    int b = idx >> 7, hd = idx & 127;
    float cv = c0[(size_t)(b0 + b) * HD_ + hd];
    float hv = h0[(size_t)(b0 + b) * HD_ + hd];
    c32[b][hd] = cv;
    c16[b][hd] = (_Float16)cv;
    xh16[b][I_ + hd] = (_Float16)hv;
  }
  if (tid < BT) ystate[tid] = y0[b0 + tid];
  __syncthreads();

  const int kb = (lane < 16) ? 0 : 8;   // A-layout half-wave K base
  const int mb = (lane < 16) ? 0 : 8;   // C/D-layout half-wave M base

  for (int t = 0; t < T_; ++t) {
    // warm caches for the next step's per-step weight tiles (global_prefetch_b8)
    if (t + 1 < T_) {
      __builtin_prefetch(WaP + (size_t)(t + 1) * (8 * 512) + (size_t)tid * 16, 0, 0);
      __builtin_prefetch(UaP + (size_t)(t + 1) * (16 * 512) + (size_t)tid * 32, 0, 0);
    }

    // ---- Phase 1: cU[16,64] = c16[16,128] @ Ua_t[128,64] (waves 0..3) ----
    if (w < 4) {
      const int nt = w;
      v8f acc = {0, 0, 0, 0, 0, 0, 0, 0};
#pragma unroll
      for (int kc = 0; kc < 4; ++kc) {
        v16h a  = ld_a16(&c16[lane & 15][kb] + kc * 32);
        v16h bm = ld_b16(UaP + ((size_t)(((t * 4 + kc) * 4 + nt) * 32 + lane)) * 16);
        acc = wmma16(a, bm, acc);
      }
      const int n = nt * 16 + (lane & 15);
#pragma unroll
      for (int j = 0; j < 8; ++j) cu[mb + j][n] = acc[j];
    }
    __syncthreads();

    // ---- Phase 2: attention logits e[b][s] (never materialize score) ----
    {
      // per-lane, per-step constants: hoisted out of the 6-unit loop
      float ba_n[4], va_n[4];
#pragma unroll
      for (int nt = 0; nt < 4; ++nt) {
        const int n = nt * 16 + (lane & 15);
        ba_n[nt] = ba[t * N_ + n];
        va_n[nt] = Va[n * T_ + t];
      }
      for (int u = w; u < BT * 6; u += 16) {      // 6 s-tiles per batch
        const int b = u / 6, st = u % 6;
        const _Float16* Arow = HhWg + (size_t)(b * S_ + st * 16) * I_ + kb;
        v16h a0 = ld_a16(Arow);
        v16h a1 = ld_a16(Arow + 32);
        float epart[8] = {0, 0, 0, 0, 0, 0, 0, 0};
#pragma unroll
        for (int nt = 0; nt < 4; ++nt) {
          v8f acc = {0, 0, 0, 0, 0, 0, 0, 0};
          acc = wmma16(a0, ld_b16(WaP + ((size_t)(((t * 2 + 0) * 4 + nt) * 32 + lane)) * 16), acc);
          acc = wmma16(a1, ld_b16(WaP + ((size_t)(((t * 2 + 1) * 4 + nt) * 32 + lane)) * 16), acc);
          const int n     = nt * 16 + (lane & 15);
          const float add = cu[b][n] + ba_n[nt];
          const float vav = va_n[nt];
#pragma unroll
          for (int j = 0; j < 8; ++j)
            epart[j] += fast_tanh(acc[j] + add) * vav;
        }
        // reduce over the 16 lanes sharing the same output row
#pragma unroll
        for (int j = 0; j < 8; ++j) {
          float v = epart[j];
          v += __shfl_xor(v, 8, 16);
          v += __shfl_xor(v, 4, 16);
          v += __shfl_xor(v, 2, 16);
          v += __shfl_xor(v, 1, 16);
          if ((lane & 15) == 0) e_l[b][st * 16 + mb + j] = v;
        }
      }
    }
    __syncthreads();

    // ---- Phase 3: softmax over S=96 (one wave per batch row) ----
    {
      const int b = w;
      float l0 = e_l[b][lane]      * SCALE;
      float l1 = e_l[b][lane + 32] * SCALE;
      float l2 = e_l[b][lane + 64] * SCALE;
      float mx = fmaxf(l0, fmaxf(l1, l2));
      for (int m = 16; m >= 1; m >>= 1) mx = fmaxf(mx, __shfl_xor(mx, m, 32));
      float p0 = __expf(l0 - mx), p1 = __expf(l1 - mx), p2 = __expf(l2 - mx);
      float s = p0 + p1 + p2;
      for (int m = 16; m >= 1; m >>= 1) s += __shfl_xor(s, m, 32);
      float inv = 1.0f / s;
      e_l[b][lane]      = p0 * inv;
      e_l[b][lane + 32] = p1 * inv;
      e_l[b][lane + 64] = p2 * inv;
    }
    __syncthreads();

    // ---- Phase 4: ctx[b,i] = beta[b,:] @ H[b,:,i]  (f32) ----
    for (int idx = tid; idx < BT * I_; idx += 512) {
      const int b = idx >> 6, i = idx & 63;
      const float* Hb = HWg + (size_t)b * S_ * I_ + i;
      float s = 0.0f;
#pragma unroll 4
      for (int ss = 0; ss < S_; ++ss) s += e_l[b][ss] * Hb[ss * I_];
      xh16[b][i] = (_Float16)s;
    }
    if (tid < BT) yacc[tid] = fcb[t];
    __syncthreads();

    // ---- Phase 5: gates[16,512] = [ctx|h][16,192] @ [W;U][192,512] ----
#pragma unroll
    for (int rep = 0; rep < 2; ++rep) {
      const int nt = w + rep * 16;                 // 0..31
      v8f acc = {0, 0, 0, 0, 0, 0, 0, 0};
      const _Float16* Arow = &xh16[lane & 15][kb];
#pragma unroll
      for (int kc = 0; kc < 6; ++kc) {
        v16h a  = ld_a16(Arow + kc * 32);
        v16h bm = ld_b16(WUP + ((size_t)((kc * 32 + nt) * 32 + lane)) * 16);
        acc = wmma16(a, bm, acc);
      }
      const int n    = nt * 16 + (lane & 15);
      const float bi = bias[n];
      const float wy = WyRow[n];                   // y_prev coefficient
#pragma unroll
      for (int j = 0; j < 8; ++j)
        gates[mb + j][n] = acc[j] + bi + ystate[mb + j] * wy;
    }
    __syncthreads();

    // ---- Phase 6: LSTM update + outputs ----
    for (int idx = tid; idx < BT * HD_; idx += 512) {
      const int b = idx >> 7, hd = idx & 127;
      const float ig = gates[b][hd];
      const float fg = gates[b][HD_ + hd];
      const float gg = gates[b][2 * HD_ + hd];
      const float og = gates[b][3 * HD_ + hd];
      const float cn = sigm(fg) * c32[b][hd] + sigm(ig) * fast_tanh(gg);
      const float hn = sigm(og) * fast_tanh(cn);
      c32[b][hd] = cn;
      c16[b][hd] = (_Float16)cn;
      xh16[b][I_ + hd] = (_Float16)hn;
      hidWg[((size_t)b * T_ + t) * HD_ + hd] = hn;
      atomicAdd(&yacc[b], hn * fcw[t * HD_ + hd]);
    }
    __syncthreads();

    if (tid < BT) {
      const float y = yacc[tid];
      ystate[tid] = y;
      y_out[(size_t)(b0 + tid) * T_ + t] = y;
    }
    __syncthreads();
  }
}

// ---------------- host launcher ----------------

extern "C" void kernel_launch(void* const* d_in, const int* in_sizes, int n_in,
                              void* d_out, int out_size, void* d_ws, size_t ws_size,
                              hipStream_t stream) {
  const float* H    = (const float*)d_in[0];
  const float* y0   = (const float*)d_in[1];
  const float* h0   = (const float*)d_in[2];
  const float* c0   = (const float*)d_in[3];
  const float* Wa   = (const float*)d_in[4];
  const float* Ua   = (const float*)d_in[5];
  const float* ba   = (const float*)d_in[6];
  const float* Va   = (const float*)d_in[7];
  const float* W    = (const float*)d_in[8];
  const float* U    = (const float*)d_in[9];
  const float* bias = (const float*)d_in[10];
  const float* fcw  = (const float*)d_in[11];
  const float* fcb  = (const float*)d_in[12];

  char* ws = (char*)d_ws;
  _Float16* Hh  = (_Float16*)(ws);                 // 4096*96*64*2   = 50331648 B
  _Float16* WaP = (_Float16*)(ws + 50331648);      // 24*64*64*2    =   196608 B
  _Float16* UaP = (_Float16*)(ws + 50528256);      // 24*128*64*2   =   393216 B
  _Float16* WUP = (_Float16*)(ws + 50921472);      // 192*512*2     =   196608 B

  float* y_out   = (float*)d_out;                  // [B,T]
  float* hid_out = y_out + (size_t)B_ * T_;        // [B,T,HD]

  convert_h_kernel<<<2048, 256, 0, stream>>>(H, Hh, B_ * S_ * I_);
  pack_wa_kernel<<<384, 256, 0, stream>>>(Wa, WaP);
  pack_ua_kernel<<<768, 256, 0, stream>>>(Ua, UaP);
  pack_wu_kernel<<<384, 256, 0, stream>>>(W, U, WUP);

  temporal_lstm_kernel<<<B_ / BT, 512, 0, stream>>>(
      H, y0, h0, c0, ba, Va, W, bias, fcw, fcb,
      Hh, WaP, UaP, WUP, y_out, hid_out);
}
